// RNN_theta_50534585205211
// MI455X (gfx1250) — matrix operand order repaired and checked by reading, not compiled, not run on hardware
//
#include <hip/hip_runtime.h>

// ---------------- problem constants ----------------
#define HDIM   2144            // N_THETA + (N_THETA+1)*N_THETA/2
#define TSTEPS 16384
#define KAUG   2272            // 128 + HDIM = 71 * 32
#define KSTEP  71              // WMMA k-steps of 32
#define UT     134             // HDIM/16 unit tiles per gate
#define NTILES 402             // 3*UT gate tiles of 16 rows
#define NWG    16              // one cluster worth of workgroups
#define TPB    864             // 27 wave32s: exactly max tiles(9) * roles(3); no idle waves, no spills

// tile storage: KSTEP * 512 bf16 elements per tile
#define TILE_ELEMS (KSTEP * 512)            // 36352
#define WAUG_ELEMS (NTILES * TILE_ELEMS)    // 14,613,504

// ---------------- workspace layout (bytes) ----------------
#define OFF_WAUG 0u
#define OFF_YBF  29227008u     // WAUG_ELEMS*2
#define OFF_HF   33421312u     // + T*64*4
#define OFF_HBF  33429888u     // + HDIM*4
#define OFF_CNT  33434176u     // + (HDIM/2)*4 (padded)

typedef __attribute__((ext_vector_type(16))) __bf16 v16bf;
typedef __attribute__((ext_vector_type(8)))  float  v8f;

union BCast { uint4 u[2]; v16bf v; };

#if __has_builtin(__builtin_amdgcn_global_load_async_to_lds_b32)
#define HAVE_ASYNC_LDS 1
typedef __attribute__((address_space(1))) int GINT;   // global-AS int
typedef __attribute__((address_space(3))) int LINT;   // LDS-AS int
#endif

__device__ __forceinline__ unsigned short f2bf(float f) {
  unsigned x = __builtin_bit_cast(unsigned, f);
  unsigned r = x + 0x7fffu + ((x >> 16) & 1u);   // round-to-nearest-even
  return (unsigned short)(r >> 16);
}

// ---------------- kernel 0: pack augmented weights into WMMA A-operand layout ----------------
// Element (tile t, kstep s, lane, j) stored at t*36352 + s*512 + lane*16 + j so the scan
// kernel's per-lane load is 32 contiguous bytes. K mapping is contiguous per lane half, matching
// the B operand staged from LDS: K = j + (lane<16 ? 0 : 16).
__global__ void pack_waug_kernel(const float* __restrict__ wih,
                                 const float* __restrict__ whh,
                                 unsigned short* __restrict__ waug) {
  int idx = blockIdx.x * 256 + threadIdx.x;
  if (idx >= WAUG_ELEMS) return;
  int t    = idx / TILE_ELEMS;
  int rem  = idx % TILE_ELEMS;
  int s    = rem >> 9;
  int e    = rem & 511;
  int lane = e >> 4;
  int j    = e & 15;
  int K    = j + (lane < 16 ? 0 : 16);
  int kk   = s * 32 + K;
  int g    = t / UT;
  int u    = t % UT;
  int row  = g * HDIM + u * 16 + (lane & 15);
  float v  = (kk < 128) ? wih[row * 128 + kk]
                        : whh[(size_t)row * HDIM + (kk - 128)];
  waug[idx] = f2bf(v);
}

// ---------------- kernel 1: y_meas -> packed bf16 pairs ----------------
__global__ void pack_y_kernel(const float* __restrict__ y, unsigned* __restrict__ ybf) {
  int idx = blockIdx.x * 256 + threadIdx.x;          // T*64
  if (idx >= TSTEPS * 64) return;
  int t = idx >> 6, p = idx & 63;
  const float* yr = y + t * 128 + p * 2;
  unsigned lo = f2bf(yr[0]), hi = f2bf(yr[1]);
  ybf[idx] = lo | (hi << 16);
}

// ---------------- kernel 2: zero state + barrier counters ----------------
__global__ void init_state_kernel(float* __restrict__ hf, unsigned* __restrict__ hbf,
                                  unsigned* __restrict__ cnt) {
  int idx = blockIdx.x * 1024 + threadIdx.x;
  if (idx < HDIM)              hf[idx] = 0.0f;
  if (idx < HDIM / 2)          hbf[idx] = 0u;
  if (idx < 64)                cnt[idx] = 0u;
}

// ---------------- kernel 3: persistent cluster-synchronized GRU scan ----------------
__global__ void __launch_bounds__(TPB)
scan_kernel(const unsigned short* __restrict__ waug,
            const unsigned* __restrict__ ybf,
            const float* __restrict__ bias,
            const float* __restrict__ bias_n,
            float* __restrict__ h_f32,
            unsigned* __restrict__ hbf,
            unsigned* __restrict__ cnt) {
  __shared__ __align__(16) unsigned x_pairs[KAUG / 2];   // 1136 packed bf16 pairs
  __shared__ float g_r[144], g_z[144], g_ign[144], g_hgn[144], h_new_l[144];

  const int tid  = threadIdx.x;
  const int wg   = blockIdx.x;
  const int wid  = tid >> 5;
  const int lane = tid & 31;
  const int half = lane >> 4;
  const int ntiles = (149 - wg) >> 4;         // 9 for wg<6, else 8 (sum = 134)

  // wave -> (unit tile, gate role) mapping; r/z/n of one unit tile share this WG
  const int  ltile   = wid / 3;
  const int  role    = wid % 3;               // 0=r, 1=z, 2=n
  const bool wactive = (ltile < ntiles);
  const int  u       = wg + (ltile << 4);
  const int  t_idx   = role * UT + u;
  const uint4* aptr  = ((const uint4*)waug) + (size_t)t_idx * (TILE_ELEMS / 8) + lane * 2;

  // per-thread gate-fusion setup (biases hoisted out of the scan loop)
  const bool uactive = (tid < 16 * ntiles);
  int   ui = 0;
  float b_r = 0.f, b_z = 0.f, b_ig = 0.f, b_n = 0.f;
  if (uactive) {
    int lt = tid >> 4, pos = tid & 15;
    ui   = (wg + (lt << 4)) * 16 + pos;
    b_r  = bias[ui];
    b_z  = bias[HDIM + ui];
    b_ig = bias[2 * HDIM + ui];
    b_n  = bias_n[ui];
  }

  // warm this wave's resident weight tile into cache (global_prefetch_b8)
  if (wactive) {
    const char* tp = (const char*)(((const uint4*)waug) + (size_t)t_idx * (TILE_ELEMS / 8));
    for (int p = lane; p < TILE_ELEMS * 2 / 128; p += 32)
      __builtin_prefetch(tp + p * 128, 0, 3);
  }

  for (int t = 0; t < TSTEPS; ++t) {
    // stage x = [y_t ; h_{t-1}] as packed bf16 pairs into LDS
#if defined(HAVE_ASYNC_LDS)
    // CDNA5 async copy: global -> LDS direct, tracked by ASYNCcnt (no VGPR round-trip)
    for (int p = tid; p < 64; p += TPB)
      __builtin_amdgcn_global_load_async_to_lds_b32(
          (GINT*)(ybf + t * 64 + p), (LINT*)&x_pairs[p], 0, 0);
    for (int p = 64 + tid; p < KAUG / 2; p += TPB)
      __builtin_amdgcn_global_load_async_to_lds_b32(
          (GINT*)(hbf + (p - 64)), (LINT*)&x_pairs[p], 0, 0);
#if __has_builtin(__builtin_amdgcn_s_wait_asynccnt)
    __builtin_amdgcn_s_wait_asynccnt(0);
#endif
#else
    for (int p = tid; p < KAUG / 2; p += TPB)
      x_pairs[p] = (p < 64) ? ybf[t * 64 + p] : hbf[p - 64];
#endif
    __syncthreads();

    v8f c0  = {0,0,0,0,0,0,0,0};   // k < 128 part (input gates)
    v8f c1a = {0,0,0,0,0,0,0,0};   // k >= 128 part, even steps
    v8f c1b = {0,0,0,0,0,0,0,0};   // k >= 128 part, odd steps
    if (wactive) {
      const unsigned* xb = x_pairs + half * 8;
      // input-side chain: k = 0..127 (4 k-steps)
      #pragma unroll
      for (int s = 0; s < 4; ++s) {
        BCast a, b;
        a.u[0] = aptr[s * 64];
        a.u[1] = aptr[s * 64 + 1];
        b.u[0] = *((const uint4*)(xb + s * 16));
        b.u[1] = *((const uint4*)(xb + s * 16 + 4));
        c0 = __builtin_amdgcn_wmma_f32_16x16x32_bf16(false, a.v, false, b.v,
                                                     (short)0, c0, false, false);
      }
      // hidden-side chain: two independent accumulator streams for pipelining
      for (int s = 4; s + 1 < KSTEP; s += 2) {
        BCast a0, b0, a1, b1;
        a0.u[0] = aptr[s * 64];
        a0.u[1] = aptr[s * 64 + 1];
        b0.u[0] = *((const uint4*)(xb + s * 16));
        b0.u[1] = *((const uint4*)(xb + s * 16 + 4));
        a1.u[0] = aptr[(s + 1) * 64];
        a1.u[1] = aptr[(s + 1) * 64 + 1];
        b1.u[0] = *((const uint4*)(xb + (s + 1) * 16));
        b1.u[1] = *((const uint4*)(xb + (s + 1) * 16 + 4));
        c1a = __builtin_amdgcn_wmma_f32_16x16x32_bf16(false, a0.v, false, b0.v,
                                                      (short)0, c1a, false, false);
        c1b = __builtin_amdgcn_wmma_f32_16x16x32_bf16(false, a1.v, false, b1.v,
                                                      (short)0, c1b, false, false);
      }
      {  // final k-step (s = 70)
        BCast a, b;
        a.u[0] = aptr[(KSTEP - 1) * 64];
        a.u[1] = aptr[(KSTEP - 1) * 64 + 1];
        b.u[0] = *((const uint4*)(xb + (KSTEP - 1) * 16));
        b.u[1] = *((const uint4*)(xb + (KSTEP - 1) * 16 + 4));
        c1a = __builtin_amdgcn_wmma_f32_16x16x32_bf16(false, a.v, false, b.v,
                                                      (short)0, c1a, false, false);
      }
      // D: all 16 columns identical; VGPR q holds row q (lanes 0-15) / row q+8 (16-31)
      if (lane == 0 || lane == 16) {
        int base = ltile * 16 + half * 8;
        if (role == 2) {
          #pragma unroll
          for (int q = 0; q < 8; ++q) {
            g_ign[base + q] = c0[q];
            g_hgn[base + q] = c1a[q] + c1b[q];
          }
        } else {
          float* dst = (role == 0) ? g_r : g_z;
          #pragma unroll
          for (int q = 0; q < 8; ++q) dst[base + q] = c0[q] + c1a[q] + c1b[q];
        }
      }
    }
    __syncthreads();

    // fused GRU gate update for this WG's hidden units
    if (uactive) {
      float hold = h_f32[ui];
      float r = 1.f / (1.f + __expf(-(g_r[tid] + b_r)));
      float z = 1.f / (1.f + __expf(-(g_z[tid] + b_z)));
      float n = tanhf(g_ign[tid] + b_ig + r * (g_hgn[tid] + b_n));
      float hn = n + z * (hold - n);
      h_f32[ui] = hn;
      h_new_l[tid] = hn;
    }
    __syncthreads();
    if (uactive && !(tid & 1)) {
      unsigned pk = (unsigned)f2bf(h_new_l[tid]) | ((unsigned)f2bf(h_new_l[tid + 1]) << 16);
      hbf[ui >> 1] = pk;
    }

    // per-step sync: cluster barrier (s_barrier_signal/-wait -3; NOP outside a
    // cluster dispatch) + monotone L2-atomic barrier as portable fallback
    __threadfence();
    __syncthreads();
    if (tid == 0) {
      __hip_atomic_fetch_add(cnt, 1u, __ATOMIC_RELEASE, __HIP_MEMORY_SCOPE_AGENT);
      unsigned target = (unsigned)(t + 1) * (unsigned)NWG;
      while (__hip_atomic_load(cnt, __ATOMIC_ACQUIRE, __HIP_MEMORY_SCOPE_AGENT) < target)
        __builtin_amdgcn_s_sleep(1);
    }
    __builtin_amdgcn_s_cluster_barrier();
    __syncthreads();
  }
}

// ---------------- kernel 4: out = w_lin @ h_final + b_lin (f32 precision) ----------------
__global__ void out_kernel(const float* __restrict__ w_lin, const float* __restrict__ b_lin,
                           const float* __restrict__ h, float* __restrict__ out) {
  __shared__ float red[256];
  int row = blockIdx.x;
  const float* wr = w_lin + (size_t)row * HDIM;
  float acc = 0.f;
  for (int j = threadIdx.x; j < HDIM; j += 256) acc += wr[j] * h[j];
  red[threadIdx.x] = acc;
  __syncthreads();
  for (int s = 128; s > 0; s >>= 1) {
    if (threadIdx.x < s) red[threadIdx.x] += red[threadIdx.x + s];
    __syncthreads();
  }
  if (threadIdx.x == 0) out[row] = red[0] + b_lin[row];
}

// ---------------- host ----------------
extern "C" void kernel_launch(void* const* d_in, const int* in_sizes, int n_in,
                              void* d_out, int out_size, void* d_ws, size_t ws_size,
                              hipStream_t stream) {
  const float* y      = (const float*)d_in[0];
  const float* wih    = (const float*)d_in[1];
  const float* whh    = (const float*)d_in[2];
  const float* bias   = (const float*)d_in[3];
  const float* bias_n = (const float*)d_in[4];
  const float* w_lin  = (const float*)d_in[5];
  const float* b_lin  = (const float*)d_in[6];
  float* out = (float*)d_out;

  char* ws = (char*)d_ws;
  unsigned short* waug = (unsigned short*)(ws + OFF_WAUG);
  unsigned*       ybf  = (unsigned*)(ws + OFF_YBF);
  float*          hf   = (float*)(ws + OFF_HF);
  unsigned*       hbf  = (unsigned*)(ws + OFF_HBF);
  unsigned*       cnt  = (unsigned*)(ws + OFF_CNT);

  pack_waug_kernel<<<WAUG_ELEMS / 256, 256, 0, stream>>>(wih, whh, waug);
  pack_y_kernel<<<(TSTEPS * 64) / 256, 256, 0, stream>>>(y, ybf);
  init_state_kernel<<<4, 1024, 0, stream>>>(hf, hbf, cnt);
  scan_kernel<<<NWG, TPB, 0, stream>>>(waug, ybf, bias, bias_n, hf, hbf, cnt);
  out_kernel<<<HDIM, 256, 0, stream>>>(w_lin, b_lin, hf, out);
}
